// Translator3_71030169141415
// MI455X (gfx1250) — compile-verified, matrix operand order checked by reading, not run on hardware
//
#include <hip/hip_runtime.h>

#define B_    32
#define T1    128
#define L1    543
#define C1    3
#define F1    64
#define G1    256     // 4*F1
#define KPAD1 224     // 9 + 192 = 201 -> pad to 7*32
#define T2    543
#define L2    64
#define F2    32
#define G2    128     // 4*F2
#define KPAD2 128     // 3 + 96 = 99 -> pad to 4*32
#define D_    32
#define H_    2
#define KD    16
#define FF_   64
#define S_    64
#define VOC   32000

typedef __attribute__((ext_vector_type(16))) __bf16          v16bf;
typedef __attribute__((ext_vector_type(8)))  float           v8f;
typedef __attribute__((ext_vector_type(8)))  unsigned short  us8;

union BF16Frag { v16bf v; us8 h[2]; unsigned short u[16]; };

__device__ __forceinline__ unsigned short f2bf(float f) {
  union { float f; unsigned int u; } x; x.f = f;
  unsigned int r = x.u + 0x7fffu + ((x.u >> 16) & 1u);   // round-to-nearest-even
  return (unsigned short)(r >> 16);
}

// A-fragment (16x32 bf16 tile, row-major source, leading dim lda elements).
// Lanes 0-15: rows 0-15, K = k0+{0..7, 16..23}; lanes 16-31: same rows, +8.
// The 16 elems are two contiguous 8x-u16 runs -> two 16B vector loads.
__device__ __forceinline__ v16bf load_a_frag(const unsigned short* A, int lda, int k0, int lane) {
  int m  = lane & 15;
  int kb = (lane >> 4) << 3;
  const unsigned short* p = A + (size_t)m * lda + k0 + kb;
  BF16Frag r;
  r.h[0] = *(const us8*)(p);
  r.h[1] = *(const us8*)(p + 16);
  return r.v;
}

// B-fragment from a pre-swizzled matrix: fragment order [tile][lane][16],
// so each lane reads 32 contiguous bytes (two 16B vector loads).
__device__ __forceinline__ v16bf load_frag_swz(const unsigned short* Bs, int tile, int lane) {
  const us8* p = (const us8*)(Bs + (((size_t)tile << 5) + lane) * 16);
  BF16Frag r;
  r.h[0] = p[0];
  r.h[1] = p[1];
  return r.v;
}

__device__ __forceinline__ float sigf(float z) { return 1.f / (1.f + __expf(-z)); }

// ---------------------------------------------------------------------------
// Pack weights to bf16 GEMM-B matrices in WMMA *fragment* layout:
//   tile (kc, nt) -> 32 lanes x 16 bf16 contiguous.
//   lane: n = nt*16 + (lane&15), kb = (lane>>4)*8
//   elem j: K = kc*32 + kb + (j<8 ? j : j+8)
// B1s: 7x16 tiles over [KPAD1][256] im2col of (Wx1 | Wh1)
// B2s: 4x8  tiles over [KPAD2][128] im2col of (Wx2 | Wh2)
// Wds: 2000 tiles over [32][32000] of Wd
// ---------------------------------------------------------------------------
__global__ void pack_weights(const float* __restrict__ Wx1, const float* __restrict__ Wh1,
                             const float* __restrict__ Wx2, const float* __restrict__ Wh2,
                             const float* __restrict__ Wd,
                             unsigned short* __restrict__ B1s,
                             unsigned short* __restrict__ B2s,
                             unsigned short* __restrict__ Wds) {
  int i = blockIdx.x * blockDim.x + threadIdx.x;
  const int N1 = KPAD1 * G1;      // 57344  (112 tiles * 512)
  const int N2 = KPAD2 * G2;      // 16384  (32 tiles * 512)
  const int N3 = D_ * VOC;        // 1024000 (2000 tiles * 512)
  if (i < N1) {
    int tile = i >> 9, r = i & 511;
    int lane = r >> 4, j = r & 15;
    int kc = tile >> 4, nt = tile & 15;
    int n  = nt * 16 + (lane & 15);
    int kb = (lane >> 4) << 3;
    int kk = kc * 32 + kb + (j < 8 ? j : j + 8);
    float v = 0.f;
    if (kk < 9)        v = Wx1[kk * G1 + n];
    else if (kk < 201) v = Wh1[(kk - 9) * G1 + n];
    B1s[i] = f2bf(v);
  } else if (i < N1 + N2) {
    int i2 = i - N1;
    int tile = i2 >> 9, r = i2 & 511;
    int lane = r >> 4, j = r & 15;
    int kc = tile >> 3, nt = tile & 7;
    int n  = nt * 16 + (lane & 15);
    int kb = (lane >> 4) << 3;
    int kk = kc * 32 + kb + (j < 8 ? j : j + 8);
    float v = 0.f;
    if (kk < 3)       v = Wx2[kk * G2 + n];
    else if (kk < 99) v = Wh2[(kk - 3) * G2 + n];
    B2s[i2] = f2bf(v);
  } else if (i < N1 + N2 + N3) {
    int i3 = i - N1 - N2;
    int tile = i3 >> 9, r = i3 & 511;
    int lane = r >> 4, j = r & 15;
    int n  = tile * 16 + (lane & 15);
    int kb = (lane >> 4) << 3;
    int kk = kb + (j < 8 ? j : j + 8);        // < 32 always
    Wds[i3] = f2bf(Wd[kk * VOC + n]);
  }
}

// ---------------------------------------------------------------------------
// ConvLSTM1D #1, one timestep: z = [x_t patch | h patch] * B1 + bias, gates
// fused in registers. Block = 4 waves; wave w owns gate-quadruple N-tiles
// {w, w+4, w+8, w+12} so each lane holds z_i/z_f/z_g/z_o for one channel.
// ---------------------------------------------------------------------------
__global__ void clstm1_step(const float* __restrict__ x,          // (B,T1,L1,C1)
                            const unsigned short* __restrict__ B1s,
                            const float* __restrict__ bias,       // (256)
                            const float* __restrict__ h_in,       // (B,L1,F1)
                            float* __restrict__ h_out,            // (B,L1,F1)
                            float* __restrict__ c_st,             // (B,L1,F1) in/out
                            int t) {
  __shared__ __align__(16) unsigned short Ash[16 * KPAD1];   // 7 KB
  const int tid = threadIdx.x;
  const int mt  = blockIdx.x;
  // im2col A tile: 16 rows of (b,l), 224 K entries (9 input taps + 192 recurrent)
  for (int e = tid; e < 16 * KPAD1; e += blockDim.x) {
    int row = e / KPAD1, kk = e - row * KPAD1;
    int r = mt * 16 + row;
    int b = r / L1, l = r - b * L1;
    float v = 0.f;
    if (kk < 9) {
      int k = kk / 3, c = kk - k * 3, ll = l + k - 1;
      if (ll >= 0 && ll < L1) v = x[(((size_t)b * T1 + t) * L1 + ll) * C1 + c];
    } else if (kk < 201) {
      int j = kk - 9, k = j >> 6, f = j & 63, ll = l + k - 1;
      if (ll >= 0 && ll < L1) v = h_in[((size_t)b * L1 + ll) * F1 + f];
    }
    Ash[e] = f2bf(v);
  }
  __syncthreads();
  const int wave = tid >> 5, lane = tid & 31;
  v8f acc[4] = {};          // N-tiles wave+0/4/8/12 -> gates i,f,g,o
  for (int kc = 0; kc < KPAD1 / 32; ++kc) {
    v16bf a = load_a_frag(Ash, KPAD1, kc * 32, lane);
#pragma unroll
    for (int q = 0; q < 4; ++q) {
      v16bf bb = load_frag_swz(B1s, kc * 16 + (wave + 4 * q), lane);
      acc[q] = __builtin_amdgcn_wmma_f32_16x16x32_bf16(false, a, false, bb,
                                                       (short)0, acc[q], false, false);
    }
  }
  const int f = wave * 16 + (lane & 15);       // filter channel 0..63
  const float gbi = bias[f], gbf = bias[64 + f], gbg = bias[128 + f], gbo = bias[192 + f];
#pragma unroll
  for (int r8 = 0; r8 < 8; ++r8) {
    int row = r8 + 8 * (lane >> 4);
    int r = mt * 16 + row;
    int b = r / L1, l = r - b * L1;
    float zi = acc[0][r8] + gbi;
    float zf = acc[1][r8] + gbf;
    float zg = acc[2][r8] + gbg;
    float zo = acc[3][r8] + gbo;
    float ig = sigf(zi), fg = sigf(zf), og = sigf(zo);
    float g  = zg > 0.f ? zg : 0.f;
    size_t idx = ((size_t)b * L1 + l) * F1 + f;
    float cn = fg * c_st[idx] + ig * g;
    c_st[idx]  = cn;
    h_out[idx] = og * (cn > 0.f ? cn : 0.f);
  }
}

// ---------------------------------------------------------------------------
// ConvLSTM1D #2, persistent: one block per batch element, 543 internal steps.
// h2/c2 state and the swizzled weights live in LDS. 8 waves: wave w handles
// M-tile (w&3) and gate-quadruple N-tiles {ft, ft+2, ft+4, ft+6}, ft = w>>2.
// ---------------------------------------------------------------------------
__global__ void clstm2_run(const float* __restrict__ h1,     // (B,543,64) = x2
                           const unsigned short* __restrict__ B2s,
                           const float* __restrict__ bias,   // (128)
                           float* __restrict__ x3out) {      // (B,64,32)
  __shared__ float h2[L2][F2];                               // 8 KB
  __shared__ float c2[L2][F2];                               // 8 KB
  __shared__ __align__(16) unsigned short Ash[L2 * KPAD2];   // 16 KB
  __shared__ __align__(16) unsigned short Bsh[KPAD2 * G2];   // 32 KB, reused 543x
  const int b = blockIdx.x, tid = threadIdx.x;
  for (int e = tid; e < L2 * F2; e += blockDim.x) { (&h2[0][0])[e] = 0.f; (&c2[0][0])[e] = 0.f; }
  for (int e = tid; e < (KPAD2 * G2) / 8; e += blockDim.x)
    ((us8*)Bsh)[e] = ((const us8*)B2s)[e];
  __syncthreads();
  const int wave = tid >> 5, lane = tid & 31;
  const int mt = wave & 3, ft = wave >> 2;
  const int f = ft * 16 + (lane & 15);         // channel 0..31
  const float gbi = bias[f], gbf = bias[32 + f], gbg = bias[64 + f], gbo = bias[96 + f];
  for (int t = 0; t < T2; ++t) {
    for (int e = tid; e < L2 * KPAD2; e += blockDim.x) {
      int l = e >> 7, kk = e & 127;
      float v = 0.f;
      if (kk < 3) {
        int ll = l + kk - 1;
        if (ll >= 0 && ll < L2) v = h1[((size_t)b * T2 + t) * L2 + ll];
      } else if (kk < 99) {
        int j = kk - 3, k = j >> 5, ff = j & 31, ll = l + k - 1;
        if (ll >= 0 && ll < L2) v = h2[ll][ff];
      }
      Ash[e] = f2bf(v);
    }
    __syncthreads();
    v8f acc[4] = {};
    for (int kc = 0; kc < KPAD2 / 32; ++kc) {
      v16bf a = load_a_frag(Ash + (size_t)mt * 16 * KPAD2, KPAD2, kc * 32, lane);
#pragma unroll
      for (int q = 0; q < 4; ++q) {
        v16bf bb = load_frag_swz(Bsh, kc * 8 + (2 * q + ft), lane);
        acc[q] = __builtin_amdgcn_wmma_f32_16x16x32_bf16(false, a, false, bb,
                                                         (short)0, acc[q], false, false);
      }
    }
#pragma unroll
    for (int r8 = 0; r8 < 8; ++r8) {
      int row = mt * 16 + r8 + 8 * (lane >> 4);
      float zi = acc[0][r8] + gbi;
      float zf = acc[1][r8] + gbf;
      float zg = acc[2][r8] + gbg;
      float zo = acc[3][r8] + gbo;
      float ig = sigf(zi), fg = sigf(zf), og = sigf(zo);
      float g  = zg > 0.f ? zg : 0.f;
      float cn = fg * c2[row][f] + ig * g;
      c2[row][f] = cn;
      h2[row][f] = og * (cn > 0.f ? cn : 0.f);
    }
    __syncthreads();
  }
  for (int e = tid; e < L2 * F2; e += blockDim.x)
    x3out[(size_t)b * L2 * F2 + e] = (&h2[0][0])[e];
}

// ---------------------------------------------------------------------------
// Transformer decoder (post-norm, causal MHA, relu FFN), one block per batch.
// Tiny (S=64, D=32): plain VALU, all state in LDS.
// ---------------------------------------------------------------------------
__global__ void decoder_kernel(float* __restrict__ x3,
    const float* __restrict__ Wq, const float* __restrict__ bq,
    const float* __restrict__ Wk, const float* __restrict__ bk,
    const float* __restrict__ Wv, const float* __restrict__ bv,
    const float* __restrict__ Wo, const float* __restrict__ bo,
    const float* __restrict__ g1, const float* __restrict__ be1,
    const float* __restrict__ W1, const float* __restrict__ bf1,
    const float* __restrict__ W2, const float* __restrict__ bf2,
    const float* __restrict__ g2, const float* __restrict__ be2) {
  __shared__ float X[S_][D_];
  __shared__ float Q[S_][H_ * KD];
  __shared__ float Km[S_][H_ * KD];
  __shared__ float V[S_][H_ * KD];
  __shared__ float U[S_][H_ * KD];
  __shared__ float O[S_][D_];
  __shared__ float Fh[S_][FF_];
  const int b = blockIdx.x, tid = threadIdx.x;
  float* xb = x3 + (size_t)b * S_ * D_;
  for (int e = tid; e < S_ * D_; e += blockDim.x) X[e / D_][e % D_] = xb[e];
  __syncthreads();
  // QKV projections: Wq/Wk/Wv flat (D, H*KD) with hk = h*16+kd
  for (int e = tid; e < S_ * 32; e += blockDim.x) {
    int s = e >> 5, hk = e & 31;
    float aq = bq[hk], ak = bk[hk], av = bv[hk];
    for (int d = 0; d < D_; ++d) {
      float xv = X[s][d];
      aq += xv * Wq[d * 32 + hk];
      ak += xv * Wk[d * 32 + hk];
      av += xv * Wv[d * 32 + hk];
    }
    Q[s][hk] = aq; Km[s][hk] = ak; V[s][hk] = av;
  }
  __syncthreads();
  // causal attention, fused scores/softmax/weighted-sum per (head, query)
  if (tid < H_ * S_) {
    int h = tid >> 6, s = tid & 63;
    float p[S_];
    float mx = -3.0e38f;
    for (int tt = 0; tt <= s; ++tt) {
      float d = 0.f;
      for (int k = 0; k < KD; ++k) d += Q[s][h * KD + k] * Km[tt][h * KD + k];
      d *= 0.25f;                 // 1/sqrt(16)
      p[tt] = d;
      mx = fmaxf(mx, d);
    }
    float sum = 0.f;
    for (int tt = 0; tt <= s; ++tt) { float e2 = __expf(p[tt] - mx); p[tt] = e2; sum += e2; }
    float inv = 1.f / sum;
    for (int k = 0; k < KD; ++k) {
      float a = 0.f;
      for (int tt = 0; tt <= s; ++tt) a += p[tt] * V[tt][h * KD + k];
      U[s][h * KD + k] = a * inv;
    }
  }
  __syncthreads();
  // output projection: Wo flat (H*KD, D)
  for (int e = tid; e < S_ * D_; e += blockDim.x) {
    int s = e >> 5, d = e & 31;
    float a = bo[d];
    for (int hk = 0; hk < 32; ++hk) a += U[s][hk] * Wo[hk * D_ + d];
    O[s][d] = a;
  }
  __syncthreads();
  if (tid < S_) {                 // residual + LN1
    int s = tid;
    float y[D_], mu = 0.f;
    for (int d = 0; d < D_; ++d) { y[d] = X[s][d] + O[s][d]; mu += y[d]; }
    mu *= (1.f / D_);
    float var = 0.f;
    for (int d = 0; d < D_; ++d) { float tv = y[d] - mu; var += tv * tv; }
    var *= (1.f / D_);
    float rs = rsqrtf(var + 1e-5f);
    for (int d = 0; d < D_; ++d) X[s][d] = (y[d] - mu) * rs * g1[d] + be1[d];
  }
  __syncthreads();
  for (int e = tid; e < S_ * FF_; e += blockDim.x) {   // FFN up + relu
    int s = e >> 6, j = e & 63;
    float a = bf1[j];
    for (int d = 0; d < D_; ++d) a += X[s][d] * W1[d * FF_ + j];
    Fh[s][j] = a > 0.f ? a : 0.f;
  }
  __syncthreads();
  for (int e = tid; e < S_ * D_; e += blockDim.x) {    // FFN down
    int s = e >> 5, d = e & 31;
    float a = bf2[d];
    for (int j = 0; j < FF_; ++j) a += Fh[s][j] * W2[j * D_ + d];
    O[s][d] = a;
  }
  __syncthreads();
  if (tid < S_) {                 // residual + LN2
    int s = tid;
    float y[D_], mu = 0.f;
    for (int d = 0; d < D_; ++d) { y[d] = X[s][d] + O[s][d]; mu += y[d]; }
    mu *= (1.f / D_);
    float var = 0.f;
    for (int d = 0; d < D_; ++d) { float tv = y[d] - mu; var += tv * tv; }
    var *= (1.f / D_);
    float rs = rsqrtf(var + 1e-5f);
    for (int d = 0; d < D_; ++d) X[s][d] = (y[d] - mu) * rs * g2[d] + be2[d];
  }
  __syncthreads();
  for (int e = tid; e < S_ * D_; e += blockDim.x) xb[e] = X[e / D_][e % D_];
}

// ---------------------------------------------------------------------------
// Vocab head: (2048 x 32) @ (32 x 32000) + bd. K=32 = one bf16 WMMA k-step.
// Grid (500, 128); block = 4 waves, one 16x16 D-tile per wave. Store-bound.
// ---------------------------------------------------------------------------
__global__ void vocab_head(const float* __restrict__ x3,     // (2048,32)
                           const unsigned short* __restrict__ Wds,
                           const float* __restrict__ bd,
                           float* __restrict__ out) {        // (2048,32000)
  __shared__ __align__(16) unsigned short Ash[16 * 32];
  const int tid = threadIdx.x;
  const int m0 = blockIdx.y * 16;
  for (int e = tid; e < 16 * 32; e += blockDim.x)
    Ash[e] = f2bf(x3[(size_t)m0 * 32 + e]);
  __syncthreads();
  const int wave = tid >> 5, lane = tid & 31;
  v16bf a  = load_a_frag(Ash, 32, 0, lane);
  v16bf bb = load_frag_swz(Wds, blockIdx.x * 4 + wave, lane);
  v8f acc = {};
  acc = __builtin_amdgcn_wmma_f32_16x16x32_bf16(false, a, false, bb, (short)0, acc, false, false);
  const int n = blockIdx.x * 64 + wave * 16 + (lane & 15);
  const float bv = bd[n];
#pragma unroll
  for (int r8 = 0; r8 < 8; ++r8) {
    int row = m0 + r8 + 8 * (lane >> 4);
    out[(size_t)row * VOC + n] = acc[r8] + bv;
  }
}

// ---------------------------------------------------------------------------
extern "C" void kernel_launch(void* const* d_in, const int* in_sizes, int n_in,
                              void* d_out, int out_size, void* d_ws, size_t ws_size,
                              hipStream_t stream) {
  (void)in_sizes; (void)n_in; (void)out_size; (void)ws_size;
  const float* x   = (const float*)d_in[0];
  const float* Wx1 = (const float*)d_in[1];
  const float* Wh1 = (const float*)d_in[2];
  const float* b1  = (const float*)d_in[3];
  const float* Wx2 = (const float*)d_in[4];
  const float* Wh2 = (const float*)d_in[5];
  const float* b2  = (const float*)d_in[6];
  const float* Wq  = (const float*)d_in[7];
  const float* bq  = (const float*)d_in[8];
  const float* Wk  = (const float*)d_in[9];
  const float* bk  = (const float*)d_in[10];
  const float* Wv  = (const float*)d_in[11];
  const float* bv  = (const float*)d_in[12];
  const float* Wo  = (const float*)d_in[13];
  const float* bo  = (const float*)d_in[14];
  const float* g1  = (const float*)d_in[15];
  const float* be1 = (const float*)d_in[16];
  const float* W1  = (const float*)d_in[17];
  const float* bf1 = (const float*)d_in[18];
  const float* W2  = (const float*)d_in[19];
  const float* bf2 = (const float*)d_in[20];
  const float* g2  = (const float*)d_in[21];
  const float* be2 = (const float*)d_in[22];
  const float* Wd  = (const float*)d_in[23];
  const float* bd  = (const float*)d_in[24];

  char* ws = (char*)d_ws;
  size_t off = 0;
  auto take = [&](size_t bytes) -> void* {
    void* p = ws + off;
    off += (bytes + 255) & ~(size_t)255;
    return p;
  };
  const size_t HBYTES = (size_t)B_ * L1 * F1 * sizeof(float);   // 4.45 MB
  float* h1a = (float*)take(HBYTES);
  float* h1b = (float*)take(HBYTES);
  float* c1  = (float*)take(HBYTES);
  unsigned short* B1s = (unsigned short*)take((size_t)KPAD1 * G1 * 2);
  unsigned short* B2s = (unsigned short*)take((size_t)KPAD2 * G2 * 2);
  unsigned short* Wds = (unsigned short*)take((size_t)D_ * VOC * 2);
  float* x3 = (float*)take((size_t)B_ * S_ * D_ * sizeof(float));

  hipMemsetAsync(h1a, 0, HBYTES, stream);
  hipMemsetAsync(c1,  0, HBYTES, stream);

  const int NPACK = KPAD1 * G1 + KPAD2 * G2 + D_ * VOC;
  pack_weights<<<(NPACK + 255) / 256, 256, 0, stream>>>(Wx1, Wh1, Wx2, Wh2, Wd,
                                                        B1s, B2s, Wds);

  const int M1TILES = (B_ * L1) / 16;   // 17376/16 = 1086, exact
  for (int t = 0; t < T1; ++t) {
    const float* hin = (t & 1) ? h1b : h1a;
    float* hout      = (t & 1) ? h1a : h1b;
    clstm1_step<<<M1TILES, 128, 0, stream>>>(x, B1s, b1, hin, hout, c1, t);
  }
  // T1=128 even -> final h in h1a
  clstm2_run<<<B_, 256, 0, stream>>>(h1a, B2s, b2, x3);
  decoder_kernel<<<B_, 256, 0, stream>>>(x3, Wq, bq, Wk, bk, Wv, bv, Wo, bo,
                                         g1, be1, W1, bf1, W2, bf2, g2, be2);
  vocab_head<<<dim3(VOC / 64, (B_ * S_) / 16), 128, 0, stream>>>(x3, Wds, bd,
                                                                 (float*)d_out);
}
